// NonlocalBlock_42314017800765
// MI455X (gfx1250) — matrix operand order repaired and checked by reading, not compiled, not run on hardware
//
#include <hip/hip_runtime.h>

#define B_  4
#define C_  256
#define CI_ 128
#define N_  4096

// Toggle for the CDNA5 async global->LDS copy path (inline asm).
#define USE_ASYNC_LDS 1

typedef __attribute__((ext_vector_type(16))) __bf16 v16bf;
typedef __attribute__((ext_vector_type(8)))  float  v8f;

union BF16Frag {
    v16bf v;
    uint4 q[2];
};

__device__ __forceinline__ unsigned short f2bf(float f) {
    unsigned int u = __float_as_uint(f);
    u += 0x7FFFu + ((u >> 16) & 1u);          // round-to-nearest-even
    return (unsigned short)(u >> 16);
}

__device__ __forceinline__ unsigned int f2bf2(float lo, float hi) {
    return (unsigned int)f2bf(lo) | ((unsigned int)f2bf(hi) << 16);
}

__device__ __forceinline__ v8f vzero8() {
    v8f z;
    for (int e = 0; e < 8; ++e) z[e] = 0.0f;
    return z;
}

__device__ __forceinline__ v8f wmma_bf16(const BF16Frag& a, const BF16Frag& b, v8f c) {
    // emits v_wmma_f32_16x16x32_bf16
    return __builtin_amdgcn_wmma_f32_16x16x32_bf16(false, a.v, false, b.v,
                                                   (short)0, c, false, false);
}

#if USE_ASYNC_LDS
// 16-byte global->LDS async copy (GLOBAL_LOAD_ASYNC_TO_LDS_B128, ASYNCcnt).
__device__ __forceinline__ void async_cp16(unsigned short* lds_dst,
                                           const unsigned short* gsrc) {
    unsigned int loff = (unsigned int)(unsigned long long)
        (__attribute__((address_space(3))) unsigned short*)lds_dst;
    asm volatile("global_load_async_to_lds_b128 %0, %1, off"
                 :: "v"(loff), "v"(gsrc) : "memory");
}
__device__ __forceinline__ void async_wait0() {
    asm volatile("s_wait_asynccnt 0x0" ::: "memory");
}
#endif

// ---------------------------------------------------------------------------
// Kernel 0: zero BN partial-sum accumulators (ws poisoned; re-zero per call).
// ---------------------------------------------------------------------------
__global__ void k0_zero(float* __restrict__ sumA, float* __restrict__ sqA) {
    sumA[threadIdx.x] = 0.0f;
    sqA[threadIdx.x]  = 0.0f;
}

// ---------------------------------------------------------------------------
// Kernel 1: fused 3-way projection GEMM.
//   out[n][j] = sum_c x[b][c][n] * w[j][c],  j in [0,384): theta|phi|g
// Block: 64 token rows x 128 cols of 384; 8 waves, each 16 rows x 64 cols.
// ---------------------------------------------------------------------------
__global__ __launch_bounds__(256) void k1_proj(
    const float* __restrict__ x,
    const float* __restrict__ w_theta, const float* __restrict__ w_phi,
    const float* __restrict__ w_g,
    const float* __restrict__ b_theta, const float* __restrict__ b_phi,
    const float* __restrict__ b_g,
    unsigned short* __restrict__ Qb, unsigned short* __restrict__ Kb,
    unsigned short* __restrict__ Vb)
{
    __shared__ __align__(16) unsigned short As[64 * 40];    // [n][c], padded
    __shared__ __align__(16) unsigned short Bst[128 * 40];  // [col][k], padded

    const int tid  = threadIdx.x;
    const int cg   = blockIdx.x;            // column group of 128 (x3)
    const int n0   = blockIdx.y * 64;       // token tile
    const int b    = blockIdx.z;
    const int wave = tid >> 5, lane = tid & 31;
    const int r    = wave & 3;              // row tile within block
    const int h    = wave >> 2;             // column half (64 cols)
    const int l15  = lane & 15;
    const int kb   = (lane >> 4) * 8;
    const int khalf = (lane >> 4) * 16;

    v8f acc[4];
    for (int t = 0; t < 4; ++t) acc[t] = vzero8();

    for (int kk = 0; kk < C_; kk += 32) {
        // stage A: x[b][kk..kk+32][n0..n0+64] -> As[n][c] bf16 (transpose).
        // float4 global loads (2/thread), scalar b16 scatter into LDS.
        #pragma unroll
        for (int it = 0; it < 2; ++it) {
            int i = tid + it * 256;          // 512 float4 chunks
            int c = i >> 4, nq = i & 15;
            float4 xv = *(const float4*)&x[(size_t)(b * C_ + kk + c) * N_ + n0 + nq * 4];
            As[(nq * 4 + 0) * 40 + c] = f2bf(xv.x);
            As[(nq * 4 + 1) * 40 + c] = f2bf(xv.y);
            As[(nq * 4 + 2) * 40 + c] = f2bf(xv.z);
            As[(nq * 4 + 3) * 40 + c] = f2bf(xv.w);
        }
        // stage B: Bst[col][k] = w_proj[jj][kk+k] bf16 (contiguous in k).
        // float4 global loads (4/thread), packed uint2 LDS stores.
        #pragma unroll
        for (int it = 0; it < 4; ++it) {
            int i = tid + it * 256;          // 1024 float4 chunks
            int j = i >> 3, kq = i & 7;
            int jg = cg * 128 + j;
            int proj = jg >> 7, jj = jg & 127;
            const float* wsrc = (proj == 0) ? w_theta : (proj == 1) ? w_phi : w_g;
            float4 wv = *(const float4*)&wsrc[(size_t)jj * C_ + kk + kq * 4];
            *(uint2*)&Bst[j * 40 + kq * 4] =
                make_uint2(f2bf2(wv.x, wv.y), f2bf2(wv.z, wv.w));
        }
        __syncthreads();

        BF16Frag a;
        a.q[0] = *(const uint4*)&As[(r * 16 + l15) * 40 + kb];
        a.q[1] = *(const uint4*)&As[(r * 16 + l15) * 40 + 16 + kb];
        #pragma unroll
        for (int t = 0; t < 4; ++t) {
            int col = h * 64 + t * 16 + l15;
            BF16Frag bb;
            bb.q[0] = *(const uint4*)&Bst[col * 40 + kb];
            bb.q[1] = *(const uint4*)&Bst[col * 40 + 16 + kb];
            acc[t] = wmma_bf16(a, bb, acc[t]);
        }
        __syncthreads();
    }

    // bias + route to Q(theta)/K(phi)/V(g), stored bf16 row-major [b][n][ci]
    #pragma unroll
    for (int t = 0; t < 4; ++t) {
        int jg = cg * 128 + h * 64 + t * 16 + l15;
        int proj = jg >> 7, jj = jg & 127;
        float bias = (proj == 0) ? b_theta[jj] : (proj == 1) ? b_phi[jj] : b_g[jj];
        unsigned short* dst = (proj == 0) ? Qb : (proj == 1) ? Kb : Vb;
        #pragma unroll
        for (int v = 0; v < 8; ++v) {
            int n = n0 + r * 16 + (lane >> 4) * 8 + v;
            dst[(size_t)(b * N_ + n) * CI_ + jj] = f2bf(acc[t][v] + bias);
        }
    }
}

// ---------------------------------------------------------------------------
// Kernel 2: flash attention, N=4096 tokens, head dim Ci=128, no scaling.
// Block = 4 waves (128 thr), each wave owns 16 query rows (64 rows/block).
// Per 32-key step: 8 wmma for S = Q K^T, online softmax, 8 wmma for O += P V.
// K tile staged via async global->LDS b128 copies; V staged transposed.
// ---------------------------------------------------------------------------
__global__ __launch_bounds__(128) void k2_attn(
    const unsigned short* __restrict__ Qb, const unsigned short* __restrict__ Kb,
    const unsigned short* __restrict__ Vb, unsigned short* __restrict__ Yb)
{
    __shared__ __align__(16) unsigned short Ks[32 * 136];   // [key][c]
    __shared__ __align__(16) unsigned short Vs[128 * 40];   // [c][key]
    __shared__ __align__(16) unsigned short Ps[4 * 16 * 40];// per-wave P tile

    const int tid  = threadIdx.x;
    const int b    = blockIdx.y;
    const int n0   = blockIdx.x * 64;
    const int wave = tid >> 5, lane = tid & 31;
    const int l15  = lane & 15;
    const int kb   = (lane >> 4) * 8;
    const int khalf = (lane >> 4) * 16;
    const int q0   = n0 + wave * 16;

    const unsigned short* Qp = Qb + (size_t)b * N_ * CI_;
    const unsigned short* Kp = Kb + (size_t)b * N_ * CI_;
    const unsigned short* Vp = Vb + (size_t)b * N_ * CI_;
    unsigned short* Pw = &Ps[wave * 16 * 40];

    // Q fragments live in registers for the whole sweep (4 x 32-k chunks)
    BF16Frag qf[4];
    #pragma unroll
    for (int kc = 0; kc < 4; ++kc) {
        const unsigned short* base = Qp + (size_t)(q0 + l15) * CI_ + kc * 32;
        qf[kc].q[0] = *(const uint4*)(base + kb);
        qf[kc].q[1] = *(const uint4*)(base + 16 + kb);
    }

    v8f o[8];
    for (int cc = 0; cc < 8; ++cc) o[cc] = vzero8();
    float mrow[8], lrow[8];
    for (int v = 0; v < 8; ++v) { mrow[v] = -1e30f; lrow[v] = 0.0f; }

    for (int j = 0; j < N_; j += 32) {
        __syncthreads();
        // stage K: straight copy, 16B chunks (async global->LDS on CDNA5)
        #pragma unroll
        for (int it = 0; it < 4; ++it) {
            int i = tid + it * 128;          // 512 x 16B chunks
            int key = i >> 4, cq = i & 15;
            const unsigned short* gs = Kp + (size_t)(j + key) * CI_ + cq * 8;
            unsigned short* ld = &Ks[key * 136 + cq * 8];
#if USE_ASYNC_LDS
            async_cp16(ld, gs);
#else
            *(uint4*)ld = *(const uint4*)gs;
#endif
        }
        // stage V transposed: uint4 global loads, b16 scatter into Vs[c][key]
        #pragma unroll
        for (int it = 0; it < 4; ++it) {
            int i = tid + it * 128;
            int key = i >> 4, cq = i & 15;
            uint4 vv = *(const uint4*)(Vp + (size_t)(j + key) * CI_ + cq * 8);
            int c0 = cq * 8;
            Vs[(c0 + 0) * 40 + key] = (unsigned short)(vv.x & 0xFFFFu);
            Vs[(c0 + 1) * 40 + key] = (unsigned short)(vv.x >> 16);
            Vs[(c0 + 2) * 40 + key] = (unsigned short)(vv.y & 0xFFFFu);
            Vs[(c0 + 3) * 40 + key] = (unsigned short)(vv.y >> 16);
            Vs[(c0 + 4) * 40 + key] = (unsigned short)(vv.z & 0xFFFFu);
            Vs[(c0 + 5) * 40 + key] = (unsigned short)(vv.z >> 16);
            Vs[(c0 + 6) * 40 + key] = (unsigned short)(vv.w & 0xFFFFu);
            Vs[(c0 + 7) * 40 + key] = (unsigned short)(vv.w >> 16);
        }
        if (j + 32 < N_) {   // pull next tile toward L2 (global_prefetch_b8)
            __builtin_prefetch(Kp + (size_t)(j + 32) * CI_ + tid * 32, 0, 1);
            __builtin_prefetch(Vp + (size_t)(j + 32) * CI_ + tid * 32, 0, 1);
        }
#if USE_ASYNC_LDS
        async_wait0();
#endif
        __syncthreads();

        // S tiles: two 16x16 key tiles, K-dim = 128 in 4 chunks
        v8f s0 = vzero8(), s1 = vzero8();
        #pragma unroll
        for (int kc = 0; kc < 4; ++kc) {
            BF16Frag b0, b1;
            b0.q[0] = *(const uint4*)&Ks[l15 * 136 + kc * 32 + kb];
            b0.q[1] = *(const uint4*)&Ks[l15 * 136 + kc * 32 + 16 + kb];
            b1.q[0] = *(const uint4*)&Ks[(16 + l15) * 136 + kc * 32 + kb];
            b1.q[1] = *(const uint4*)&Ks[(16 + l15) * 136 + kc * 32 + 16 + kb];
            s0 = wmma_bf16(qf[kc], b0, s0);
            s1 = wmma_bf16(qf[kc], b1, s1);
        }

        // online softmax: row = (lane>>4)*8 + v, cols across 16 lanes
        float sc[8];
        #pragma unroll
        for (int v = 0; v < 8; ++v) {
            float mt = fmaxf(s0[v], s1[v]);
            #pragma unroll
            for (int off = 1; off < 16; off <<= 1)
                mt = fmaxf(mt, __shfl_xor(mt, off, 32));
            float nm = fmaxf(mrow[v], mt);
            sc[v] = __expf(mrow[v] - nm);
            float p0 = __expf(s0[v] - nm);
            float p1 = __expf(s1[v] - nm);
            float rs = p0 + p1;
            #pragma unroll
            for (int off = 1; off < 16; off <<= 1)
                rs += __shfl_xor(rs, off, 32);
            lrow[v] = lrow[v] * sc[v] + rs;
            mrow[v] = nm;
            int m = (lane >> 4) * 8 + v;
            Pw[m * 40 + l15]      = f2bf(p0);
            Pw[m * 40 + 16 + l15] = f2bf(p1);
        }
        #pragma unroll
        for (int cc = 0; cc < 8; ++cc)
            #pragma unroll
            for (int v = 0; v < 8; ++v)
                o[cc][v] *= sc[v];
        __syncthreads();

        // O += P V : P re-read in A-layout, V B-frags contiguous from Vs
        BF16Frag pa;
        pa.q[0] = *(const uint4*)&Pw[l15 * 40 + kb];
        pa.q[1] = *(const uint4*)&Pw[l15 * 40 + 16 + kb];
        #pragma unroll
        for (int cc = 0; cc < 8; ++cc) {
            BF16Frag bv;
            bv.q[0] = *(const uint4*)&Vs[(cc * 16 + l15) * 40 + kb];
            bv.q[1] = *(const uint4*)&Vs[(cc * 16 + l15) * 40 + 16 + kb];
            o[cc] = wmma_bf16(pa, bv, o[cc]);
        }
    }

    // normalize by l, store y^T as bf16 [b][n][ci]
    #pragma unroll
    for (int v = 0; v < 8; ++v) {
        float inv = 1.0f / lrow[v];
        int n = q0 + (lane >> 4) * 8 + v;
        #pragma unroll
        for (int cc = 0; cc < 8; ++cc)
            Yb[(size_t)(b * N_ + n) * CI_ + cc * 16 + l15] = f2bf(o[cc][v] * inv);
    }
}

// ---------------------------------------------------------------------------
// Kernel 3: output 1x1 conv  Wy[n][o] = Y[n][:]·w_out[o][:] + b_out[o]
// plus per-channel sum/sumsq partials for BatchNorm (shfl-fold + atomicAdd).
// ---------------------------------------------------------------------------
__global__ __launch_bounds__(256) void k3_outconv(
    const unsigned short* __restrict__ Yb, const float* __restrict__ w_out,
    const float* __restrict__ b_out, float* __restrict__ Wy,
    float* __restrict__ sumA, float* __restrict__ sqA)
{
    __shared__ __align__(16) unsigned short Ws[128 * 136];  // [o][ci]
    const int tid = threadIdx.x;
    const int n0 = blockIdx.x * 64, o0 = blockIdx.y * 128, b = blockIdx.z;

    // stage weights: float4 loads, packed uint2 LDS stores (native [o][ci])
    #pragma unroll
    for (int it = 0; it < 16; ++it) {
        int i = tid + it * 256;              // 4096 float4 chunks
        int o = i >> 5, kq = i & 31;
        float4 wv = *(const float4*)&w_out[(size_t)(o0 + o) * CI_ + kq * 4];
        *(uint2*)&Ws[o * 136 + kq * 4] =
            make_uint2(f2bf2(wv.x, wv.y), f2bf2(wv.z, wv.w));
    }
    __syncthreads();

    const int wave = tid >> 5, lane = tid & 31;
    const int r = wave & 3, h = wave >> 2;
    const int l15 = lane & 15;
    const int kb = (lane >> 4) * 8, khalf = (lane >> 4) * 16;
    const int rowbase = n0 + r * 16;

    v8f acc[4];
    for (int t = 0; t < 4; ++t) acc[t] = vzero8();

    #pragma unroll
    for (int kc = 0; kc < 4; ++kc) {
        BF16Frag a;
        const unsigned short* yb = Yb + (size_t)(b * N_ + rowbase + l15) * CI_ + kc * 32;
        a.q[0] = *(const uint4*)(yb + kb);
        a.q[1] = *(const uint4*)(yb + 16 + kb);
        #pragma unroll
        for (int t = 0; t < 4; ++t) {
            int ol = h * 64 + t * 16 + l15;
            BF16Frag bb;
            bb.q[0] = *(const uint4*)&Ws[ol * 136 + kc * 32 + kb];
            bb.q[1] = *(const uint4*)&Ws[ol * 136 + kc * 32 + 16 + kb];
            acc[t] = wmma_bf16(a, bb, acc[t]);
        }
    }

    #pragma unroll
    for (int t = 0; t < 4; ++t) {
        int ocol = o0 + h * 64 + t * 16 + l15;
        float bo = b_out[ocol];
        float ps = 0.0f, pq = 0.0f;
        #pragma unroll
        for (int v = 0; v < 8; ++v) {
            int n = rowbase + (lane >> 4) * 8 + v;
            float val = acc[t][v] + bo;
            Wy[(size_t)(b * N_ + n) * C_ + ocol] = val;  // [b][n][o], coalesced
            ps += val;
            pq += val * val;
        }
        ps += __shfl_xor(ps, 16, 32);
        pq += __shfl_xor(pq, 16, 32);
        if (lane < 16) {
            atomicAdd(&sumA[ocol], ps);
            atomicAdd(&sqA[ocol], pq);
        }
    }
}

// ---------------------------------------------------------------------------
// Kernel 4a: finalize BN stats -> per-channel scale/shift.
// ---------------------------------------------------------------------------
__global__ void k4a_stats(const float* __restrict__ sumA, const float* __restrict__ sqA,
                          const float* __restrict__ gamma, const float* __restrict__ beta,
                          float* __restrict__ scl, float* __restrict__ shf)
{
    int o = threadIdx.x;
    const float inv = 1.0f / (float)(B_ * N_);
    float mean = sumA[o] * inv;
    float var  = sqA[o] * inv - mean * mean;
    float s = gamma[o] * rsqrtf(var + 1e-5f);
    scl[o] = s;
    shf[o] = beta[o] - mean * s;
}

// ---------------------------------------------------------------------------
// Kernel 4b: LDS-tiled transpose [b][n][o] -> [b][o][n], BN affine + residual.
// float4 on both global sides, transpose through padded LDS tile.
// ---------------------------------------------------------------------------
__global__ __launch_bounds__(256) void k4b_final(
    const float* __restrict__ Wy, const float* __restrict__ x,
    const float* __restrict__ scl, const float* __restrict__ shf,
    float* __restrict__ out)
{
    __shared__ float tile[64 * 65];
    const int tid = threadIdx.x;
    const int o0 = blockIdx.x * 64, n0 = blockIdx.y * 64, b = blockIdx.z;

    #pragma unroll
    for (int it = 0; it < 4; ++it) {
        int i = tid + it * 256;              // 1024 float4 chunks
        int nl = i >> 4, oq = i & 15;
        float4 wv = *(const float4*)&Wy[(size_t)(b * N_ + n0 + nl) * C_ + o0 + oq * 4];
        tile[nl * 65 + oq * 4 + 0] = wv.x;
        tile[nl * 65 + oq * 4 + 1] = wv.y;
        tile[nl * 65 + oq * 4 + 2] = wv.z;
        tile[nl * 65 + oq * 4 + 3] = wv.w;
    }
    __syncthreads();
    #pragma unroll
    for (int it = 0; it < 4; ++it) {
        int i = tid + it * 256;
        int ol = i >> 4, nq = i & 15;
        int o = o0 + ol;
        float s = scl[o], t = shf[o];
        size_t idx = (size_t)(b * C_ + o) * N_ + n0 + nq * 4;
        float4 xv = *(const float4*)&x[idx];
        float4 r;
        r.x = tile[(nq * 4 + 0) * 65 + ol] * s + t + xv.x;
        r.y = tile[(nq * 4 + 1) * 65 + ol] * s + t + xv.y;
        r.z = tile[(nq * 4 + 2) * 65 + ol] * s + t + xv.z;
        r.w = tile[(nq * 4 + 3) * 65 + ol] * s + t + xv.w;
        *(float4*)&out[idx] = r;
    }
}

// ---------------------------------------------------------------------------
extern "C" void kernel_launch(void* const* d_in, const int* in_sizes, int n_in,
                              void* d_out, int out_size, void* d_ws, size_t ws_size,
                              hipStream_t stream)
{
    (void)in_sizes; (void)n_in; (void)out_size; (void)ws_size;

    const float* x       = (const float*)d_in[0];
    const float* w_g     = (const float*)d_in[1];
    const float* b_g     = (const float*)d_in[2];
    const float* w_theta = (const float*)d_in[3];
    const float* b_theta = (const float*)d_in[4];
    const float* w_phi   = (const float*)d_in[5];
    const float* b_phi   = (const float*)d_in[6];
    const float* w_out   = (const float*)d_in[7];
    const float* b_out   = (const float*)d_in[8];
    const float* gamma   = (const float*)d_in[9];
    const float* beta    = (const float*)d_in[10];
    float* out = (float*)d_out;

    // Workspace layout (Wy aliases Q/K/V which are dead after attention):
    //   [0,4M)   Yb bf16     [4M,8M)  Qb bf16   [8M,12M) Kb bf16
    //   [12M,16M) Vb bf16    [4M,20M) Wy f32    [20M,+4K) BN stats
    char* ws = (char*)d_ws;
    unsigned short* Yb = (unsigned short*)(ws + (0ull  << 20));
    unsigned short* Qb = (unsigned short*)(ws + (4ull  << 20));
    unsigned short* Kb = (unsigned short*)(ws + (8ull  << 20));
    unsigned short* Vb = (unsigned short*)(ws + (12ull << 20));
    float* Wy    = (float*)(ws + (4ull  << 20));
    float* stats = (float*)(ws + (20ull << 20));
    float* sumA = stats, * sqA = stats + 256, * scl = stats + 512, * shf = stats + 768;

    k0_zero   <<<1, 256, 0, stream>>>(sumA, sqA);
    k1_proj   <<<dim3(3, N_ / 64, B_), 256, 0, stream>>>(x, w_theta, w_phi, w_g,
                                                         b_theta, b_phi, b_g, Qb, Kb, Vb);
    k2_attn   <<<dim3(N_ / 64, B_), 128, 0, stream>>>(Qb, Kb, Vb, Yb);
    k3_outconv<<<dim3(N_ / 64, 2, B_), 256, 0, stream>>>(Yb, w_out, b_out, Wy, sumA, sqA);
    k4a_stats <<<1, 256, 0, stream>>>(sumA, sqA, gamma, beta, scl, shf);
    k4b_final <<<dim3(C_ / 64, N_ / 64, B_), 256, 0, stream>>>(Wy, x, scl, shf, out);
}
